// MultiRateMambaFusion_9122510537268
// MI455X (gfx1250) — matrix operand order repaired and checked by reading, compiled, not run on hardware
//
#include <hip/hip_runtime.h>
#include <hip/hip_bf16.h>
#include <math.h>
#include <stddef.h>

// ---------------------------------------------------------------------------
// MI455X (gfx1250) implementation, round 5 (final tuning).
// * bf16 operand storage everywhere (pre-transposed [K,N] bf16 weights,
//   bf16 activation shadows), f32 WMMA accumulate, f32 epilogues.
// * k_gemm<MF>: MF*64 x 64 tile, double-buffered LDS, staging via
//   GLOBAL_LOAD_ASYNC_TO_LDS_B128 (ASYNCcnt), ds_load_b128 A fragments,
//   chunk-XOR B swizzle, one barrier per K-step. MF=2 for big-M GEMMs
//   (8 WMMAs / 4 b128 + 64 u16 LDS loads per wave-step), MF=1 for M=4096 (occupancy).
// ---------------------------------------------------------------------------

typedef __attribute__((ext_vector_type(16))) __bf16 v16bf;
typedef __attribute__((ext_vector_type(8)))  float  v8f;
typedef __attribute__((ext_vector_type(4)))  unsigned u4x;   // 16-byte chunk
typedef __attribute__((vector_size(16)))     int     gv4i;   // builtin's elem type

#if defined(__gfx1250__) && __has_builtin(__builtin_amdgcn_global_load_async_to_lds_b128)
#define HAVE_ASYNC_LDS 1
#else
#define HAVE_ASYNC_LDS 0
#endif

#if HAVE_ASYNC_LDS
#define ASYNC_CP128(dst_lds, src_glob)                                         \
  __builtin_amdgcn_global_load_async_to_lds_b128(                              \
      (__attribute__((address_space(1))) gv4i*)(src_glob),                     \
      (__attribute__((address_space(3))) gv4i*)(dst_lds), 0, 0)
#endif

__device__ inline void async_wait0() {
#if HAVE_ASYNC_LDS
#if __has_builtin(__builtin_amdgcn_s_wait_asynccnt)
  __builtin_amdgcn_s_wait_asynccnt(0);
#else
  asm volatile("s_wait_asynccnt 0x0" ::: "memory");
#endif
#endif
}

__device__ inline unsigned short f2bf(float f) {
  unsigned u = __builtin_bit_cast(unsigned, f);
  u += 0x7FFFu + ((u >> 16) & 1u);          // round-to-nearest-even
  return (unsigned short)(u >> 16);
}
__device__ inline unsigned fenc(float f) {
  unsigned u = __builtin_bit_cast(unsigned, f);
  return (u & 0x80000000u) ? ~u : (u | 0x80000000u);
}
__device__ inline float fdec(unsigned k) {
  unsigned u = (k & 0x80000000u) ? (k ^ 0x80000000u) : ~k;
  return __builtin_bit_cast(float, u);
}
__device__ inline float sigf(float x) { return 1.f / (1.f + __expf(-x)); }

// ---------------------------------------------------------------------------
// WMMA GEMM: C[M,N] = A[M,K](bf16) * Wb[K,N](bf16) [+C] [+bias]; epilogue f32.
// flags: 1=accumulate, 2=relu, 4=sigmoid. Cbf: optional bf16 shadow of C.
// MF=1: 64-row tile (M%64==0); MF=2: 128-row tile (M%128==0).
// 128 threads = 4 waves; wave owns MF*16 rows. K % 32 == 0 required.
// ---------------------------------------------------------------------------
template <int MF>
__global__ __launch_bounds__(128) void k_gemm(
    const unsigned short* __restrict__ A, int lda,
    const unsigned short* __restrict__ Wb, int ldw,
    float* __restrict__ C, int ldc, unsigned short* __restrict__ Cbf,
    const float* __restrict__ bias,
    int M, int N, int K, int flags)
{
  __shared__ __align__(16) unsigned short As[2][MF * 64][40];  // [m][k], 80B stride
  __shared__ __align__(16) unsigned short Bs[2][32][72];       // [k][n], 144B stride
  const int tid  = threadIdx.x;
  const int wv   = tid >> 5;
  const int lane = tid & 31;
  const int half = lane >> 4;
  const int l16  = lane & 15;
  const int m0   = blockIdx.y * (MF * 64);
  const int n0   = blockIdx.x * 64;

  v8f acc[MF][4];
#pragma unroll
  for (int mf = 0; mf < MF; mf++)
#pragma unroll
    for (int i = 0; i < 4; i++) {
      v8f z = {0.f, 0.f, 0.f, 0.f, 0.f, 0.f, 0.f, 0.f};
      acc[mf][i] = z;
    }

  auto stage = [&](int pb, int kk) {
    // A tile: MF*64 rows x 32 halves; unguarded (M multiple of MF*64).
#pragma unroll
    for (int i = 0; i < 2 * MF; i++) {
      int c = i * 128 + tid;        // MF*256 chunks
      int r = c >> 2;               // 4 chunks per row
      int col8 = (c & 3) * 8;
      const unsigned short* gp = A + (size_t)(m0 + r) * lda + kk + col8;
#if HAVE_ASYNC_LDS
      ASYNC_CP128(&As[pb][r][col8], gp);
#else
      *(u4x*)&As[pb][r][col8] = *(const u4x*)gp;
#endif
    }
    // B tile: 32 rows x 64 halves, chunk-XOR swizzle on rows>=16.
#pragma unroll
    for (int i = 0; i < 2; i++) {
      int c = i * 128 + tid;        // 256 chunks
      int r = c >> 3;               // 8 chunks per row (k row)
      int col8 = (c & 7) * 8;
      int swz = (r & 16) ? 8 : 0;
      int gn0 = n0 + col8;
      const unsigned short* gp = Wb + (size_t)(kk + r) * ldw + gn0;
      if (((ldw & 7) == 0) && (gn0 + 8 <= N)) {
#if HAVE_ASYNC_LDS
        ASYNC_CP128(&Bs[pb][r][col8 ^ swz], gp);
#else
        *(u4x*)&Bs[pb][r][col8 ^ swz] = *(const u4x*)gp;
#endif
      } else {
#pragma unroll
        for (int j = 0; j < 8; j++) {
          int gn = gn0 + j;
          Bs[pb][r][(col8 ^ swz) + j] =
              (gn < N) ? Wb[(size_t)(kk + r) * ldw + gn] : (unsigned short)0;
        }
      }
    }
    // CDNA5 prefetch of the K-tile after next (global_prefetch_b8)
    if (kk + 32 < K)
      __builtin_prefetch(A + (size_t)(m0 + (tid >> 1)) * lda + kk + 32 + (tid & 1) * 16, 0, 1);
  };

  const int nIter = K >> 5;
  stage(0, 0);
  async_wait0();
  __syncthreads();

  for (int it = 0; it < nIter; it++) {
    if (it + 1 < nIter) stage((it + 1) & 1, (it + 1) << 5);
    const int pb = it & 1;

    // A fragments: lane l16 = M row; elem e -> K = (e/8)*16 + half*8 + e%8
    union Frag { v16bf v; u4x q[2]; } fa[MF];
#pragma unroll
    for (int mf = 0; mf < MF; mf++) {
      const int mrow = wv * (MF * 16) + mf * 16 + l16;
      fa[mf].q[0] = *(const u4x*)&As[pb][mrow][half * 8];
      fa[mf].q[1] = *(const u4x*)&As[pb][mrow][16 + half * 8];
    }

    const int sw = (lane & 16) ? 8 : 0;   // unused; B swizzle is undone per element via kb & 16
#pragma unroll
    for (int nt = 0; nt < 4; nt++) {
      union FragB { v16bf v; u4x q[2]; } fb;
#pragma unroll
      for (int s = 0; s < 16; s++) {
        const int kb = (s < 8) ? (half * 8 + s) : (16 + half * 8 + (s - 8));
        fb.v[s] = __builtin_bit_cast(__bf16, Bs[pb][kb][(nt * 16 + l16) ^ ((kb & 16) ? 8 : 0)]);
      }
#pragma unroll
      for (int mf = 0; mf < MF; mf++)
        acc[mf][nt] = __builtin_amdgcn_wmma_f32_16x16x32_bf16(
            false, fa[mf].v, false, fb.v, (short)0, acc[mf][nt], false, false);
    }
    async_wait0();
    __syncthreads();
  }

  // C/D layout: VGPR r -> M = r + 8*half; lane%16 -> N
#pragma unroll
  for (int mf = 0; mf < MF; mf++)
#pragma unroll
    for (int nt = 0; nt < 4; nt++)
#pragma unroll
      for (int r = 0; r < 8; r++) {
        int gm = m0 + wv * (MF * 16) + mf * 16 + r + 8 * half;
        int gn = n0 + nt * 16 + l16;
        if (gn < N) {
          float v = acc[mf][nt][r];
          if (flags & 1) v += C[(size_t)gm * ldc + gn];
          if (bias)      v += bias[gn];
          if (flags & 2) v = fmaxf(v, 0.f);
          if (flags & 4) v = sigf(v);
          C[(size_t)gm * ldc + gn] = v;
          if (Cbf) Cbf[(size_t)gm * ldc + gn] = f2bf(v);
        }
      }
}

// ---------------------------------------------------------------------------
// Utility / pointwise kernels
// ---------------------------------------------------------------------------
__global__ void k_fill(float* p, size_t n, float v) {
  size_t i = (size_t)blockIdx.x * 256 + threadIdx.x;
  if (i < n) p[i] = v;
}
__global__ void k_fillu(unsigned short* p, size_t n) {
  size_t i = (size_t)blockIdx.x * 256 + threadIdx.x;
  if (i < n) p[i] = 0;
}
__global__ void k_cvt(const float* __restrict__ in, unsigned short* __restrict__ out, size_t n) {
  size_t i = (size_t)blockIdx.x * 256 + threadIdx.x;
  if (i < n) out[i] = f2bf(in[i]);
}
__global__ void k_add_sh(float* a, const float* __restrict__ b,
                         unsigned short* __restrict__ sh, size_t n) {
  size_t i = (size_t)blockIdx.x * 256 + threadIdx.x;
  if (i < n) { float v = a[i] + b[i]; a[i] = v; sh[i] = f2bf(v); }
}
__global__ void k_wT(const float* __restrict__ W, unsigned short* __restrict__ out,
                     int Kd, int Nd, int transpose) {
  int i = blockIdx.x * 256 + threadIdx.x;
  if (i >= Kd * Nd) return;
  int k = i / Nd, n = i - k * Nd;
  float v = transpose ? W[(size_t)n * Kd + k] : W[(size_t)k * Nd + n];
  out[i] = f2bf(v);
}

__global__ void k_input_proj(const float* __restrict__ vm, const float* __restrict__ pb,
                             const float* __restrict__ qb, const float* __restrict__ va,
                             const float* __restrict__ W, const float* __restrict__ bi,
                             unsigned short* __restrict__ h0)
{
  size_t i = (size_t)blockIdx.x * 256 + threadIdx.x;   // ROWS*64
  size_t row = i >> 6;
  int o = (int)(i & 63);
  float r = W[o * 4 + 0] * vm[row] + W[o * 4 + 1] * pb[row] +
            W[o * 4 + 2] * qb[row] + W[o * 4 + 3] * va[row] + bi[o];
  h0[i] = f2bf(r);
}

__global__ void k_gat_logits(const float* __restrict__ hw, const float* __restrict__ a_s,
                             const float* __restrict__ a_d,
                             float* __restrict__ als, float* __restrict__ ald)
{
  int i = blockIdx.x * 256 + threadIdx.x;              // ROWS*4
  size_t row = (size_t)(i >> 2);
  int h = i & 3;
  const float* v = hw + row * 256 + h * 64;
  float s = 0.f, d = 0.f;
#pragma unroll
  for (int c = 0; c < 64; c++) { float x = v[c]; s += x * a_s[h * 64 + c]; d += x * a_d[h * 64 + c]; }
  als[i] = s; ald[i] = d;
}

__global__ __launch_bounds__(256) void k_gat_maxsum(
    const float* __restrict__ als, const float* __restrict__ ald,
    const int* __restrict__ src, const int* __restrict__ dst,
    float* __restrict__ mbuf, float* __restrict__ sbuf)
{
  __shared__ unsigned mk[2048];   // 512 nodes * 4 heads
  __shared__ float    ss[2048];
  const int bt = blockIdx.x, tid = threadIdx.x;
  for (int i = tid; i < 2048; i += 256) { mk[i] = 0u; ss[i] = 0.f; }
  __syncthreads();
  for (int e = tid; e < 2048; e += 256) {
    int s = src[e], d = dst[e];
#pragma unroll
    for (int h = 0; h < 4; h++) {
      float ev = als[((size_t)bt * 512 + s) * 4 + h] + ald[((size_t)bt * 512 + d) * 4 + h];
      ev = ev > 0.f ? ev : 0.2f * ev;
      atomicMax(&mk[d * 4 + h], fenc(ev));
    }
  }
  __syncthreads();
  for (int e = tid; e < 2048; e += 256) {
    int s = src[e], d = dst[e];
#pragma unroll
    for (int h = 0; h < 4; h++) {
      float ev = als[((size_t)bt * 512 + s) * 4 + h] + ald[((size_t)bt * 512 + d) * 4 + h];
      ev = ev > 0.f ? ev : 0.2f * ev;
      atomicAdd(&ss[d * 4 + h], __expf(ev - fdec(mk[d * 4 + h])));
    }
  }
  __syncthreads();
  for (int i = tid; i < 2048; i += 256) {
    mbuf[(size_t)bt * 2048 + i] = fdec(mk[i]);
    sbuf[(size_t)bt * 2048 + i] = ss[i];
  }
}

__global__ __launch_bounds__(256) void k_gat_aggregate(
    const float* __restrict__ hw, const float* __restrict__ als, const float* __restrict__ ald,
    const float* __restrict__ mbuf, const float* __restrict__ sbuf,
    const int* __restrict__ src, const int* __restrict__ dst,
    float* __restrict__ out)
{
  const int blk = blockIdx.x;
  const int bt  = blk >> 11;
  const int e   = blk & 2047;
  const int c   = threadIdx.x;
  const int h   = c >> 6;
  int s = src[e], d = dst[e];
  size_t bs = (size_t)bt * 512 + s;
  size_t bd = (size_t)bt * 512 + d;
  float ev = als[bs * 4 + h] + ald[bd * 4 + h];
  ev = ev > 0.f ? ev : 0.2f * ev;
  float alpha = __expf(ev - mbuf[bd * 4 + h]) / (sbuf[bd * 4 + h] + 1e-16f);
  atomicAdd(&out[bd * 256 + c], alpha * hw[bs * 256 + c]);
}

__global__ void k_bias_elu(float* x, const float* __restrict__ b,
                           unsigned short* __restrict__ sh) {
  size_t i = (size_t)blockIdx.x * 256 + threadIdx.x;
  int c = (int)(i & 255);
  float v = x[i] + b[c];
  v = v > 0.f ? v : (__expf(v) - 1.f);
  x[i] = v; sh[i] = f2bf(v);
}

__global__ void k_seq_tr(const float* __restrict__ in, float* __restrict__ out) {
  size_t i = (size_t)blockIdx.x * 256 + threadIdx.x;
  int c = (int)(i & 255);
  size_t row = i >> 8;
  int n = (int)(row & 511);
  int bt = (int)(row >> 9);
  int t = bt & 63, b = bt >> 6;
  out[(((size_t)b * 512 + n) * 64 + t) * 256 + c] = in[i];
}

__global__ void k_lstm_point(const float* __restrict__ gates,
                             const float* __restrict__ bih, const float* __restrict__ bhh,
                             float* __restrict__ h, unsigned short* __restrict__ hbf,
                             float* __restrict__ c, float* __restrict__ y, int t)
{
  int i = blockIdx.x * 256 + threadIdx.x;   // BN*256
  int row = i >> 8, j = i & 255;
  size_t g0 = (size_t)row * 1024;
  float gi = gates[g0 + j]        + bih[j]        + bhh[j];
  float gf = gates[g0 + 256 + j]  + bih[256 + j]  + bhh[256 + j];
  float gg = gates[g0 + 512 + j]  + bih[512 + j]  + bhh[512 + j];
  float go = gates[g0 + 768 + j]  + bih[768 + j]  + bhh[768 + j];
  float cv = sigf(gf) * c[i] + sigf(gi) * tanhf(gg);
  float hv = sigf(go) * tanhf(cv);
  c[i] = cv; h[i] = hv; hbf[i] = f2bf(hv);
  y[((size_t)row * 64 + t) * 256 + j] = hv;
}

__global__ __launch_bounds__(64) void k_mha_attn(
    const float* __restrict__ Q, const float* __restrict__ K,
    const float* __restrict__ V, unsigned short* __restrict__ O)
{
  __shared__ float Ks[64][33];
  __shared__ float Vs[64][33];
  const int bn = blockIdx.x, h = blockIdx.y;
  const int t = threadIdx.x;
  const size_t base = ((size_t)bn * 64) * 128 + h * 32;
#pragma unroll
  for (int d = 0; d < 32; d++) {
    Ks[t][d] = K[base + (size_t)t * 128 + d];
    Vs[t][d] = V[base + (size_t)t * 128 + d];
  }
  __syncthreads();
  float q[32];
#pragma unroll
  for (int d = 0; d < 32; d++) q[d] = Q[base + (size_t)t * 128 + d];
  const float scale = 0.17677669529663687f;  // 1/sqrt(32)
  float sc[64], mx = -3.0e38f;
  for (int k2 = 0; k2 < 64; k2++) {
    float s = 0.f;
#pragma unroll
    for (int d = 0; d < 32; d++) s += q[d] * Ks[k2][d];
    s *= scale; sc[k2] = s; mx = fmaxf(mx, s);
  }
  float se = 0.f;
  for (int k2 = 0; k2 < 64; k2++) { float p = __expf(sc[k2] - mx); sc[k2] = p; se += p; }
  float inv = 1.f / se;
#pragma unroll
  for (int d = 0; d < 32; d++) {
    float o = 0.f;
    for (int k2 = 0; k2 < 64; k2++) o += sc[k2] * Vs[k2][d];
    O[base + (size_t)t * 128 + d] = f2bf(o * inv);
  }
}

__global__ void k_res_ln(const float* __restrict__ a, const float* __restrict__ b,
                         const float* __restrict__ g, const float* __restrict__ be,
                         float* __restrict__ out, unsigned short* __restrict__ sh, int dim)
{
  __shared__ float red[128];
  int row = blockIdx.x, t = threadIdx.x;
  float v = a[(size_t)row * dim + t];
  if (b) v += b[(size_t)row * dim + t];
  red[t] = v; __syncthreads();
  for (int s = dim >> 1; s > 0; s >>= 1) { if (t < s) red[t] += red[t + s]; __syncthreads(); }
  float mean = red[0] / dim; __syncthreads();
  float dv = v - mean;
  red[t] = dv * dv; __syncthreads();
  for (int s = dim >> 1; s > 0; s >>= 1) { if (t < s) red[t] += red[t + s]; __syncthreads(); }
  float var = red[0] / dim;
  float o = dv * rsqrtf(var + 1e-5f) * g[t] + be[t];
  out[(size_t)row * dim + t] = o;
  if (sh) sh[(size_t)row * dim + t] = f2bf(o);
}

__global__ void k_mean_pool(const float* __restrict__ x, float* __restrict__ out) {
  int i = blockIdx.x * 128 + threadIdx.x;   // BN*128
  int bn = i >> 7, f = i & 127;
  float s = 0.f;
  for (int t = 0; t < 64; t++) s += x[((size_t)bn * 64 + t) * 128 + f];
  out[i] = s * (1.f / 64.f);
}

__global__ void k_concat2(const float* __restrict__ a, const float* __restrict__ b,
                          unsigned short* __restrict__ out) {
  int i = blockIdx.x * 256 + threadIdx.x;   // BN*256
  int bn = i >> 8, j = i & 255;
  float v = (j < 128) ? a[bn * 128 + j] : b[bn * 128 + (j - 128)];
  out[i] = f2bf(v);
}

__global__ void k_gated_cat(const float* __restrict__ fp, const float* __restrict__ sp,
                            const float* __restrict__ g, unsigned short* __restrict__ out) {
  int i = blockIdx.x * 256 + threadIdx.x;   // BN*256
  int bn = i >> 8, j = i & 255;
  float v = (j < 128) ? fp[bn * 128 + j] * g[bn * 128 + j]
                      : sp[bn * 128 + (j - 128)] * (1.f - g[bn * 128 + (j - 128)]);
  out[i] = f2bf(v);
}

__global__ void k_ef_build(const float* __restrict__ fused, const int* __restrict__ src,
                           const int* __restrict__ dst, unsigned short* __restrict__ ef) {
  size_t i = (size_t)blockIdx.x * 256 + threadIdx.x;  // B*E*256
  int c = (int)(i & 255);
  size_t row = i >> 8;
  int e = (int)(row & 2047), b = (int)(row >> 11);
  int node = (c < 128) ? src[e] : dst[e];
  ef[i] = f2bf(fused[((size_t)b * 512 + node) * 128 + (c & 127)]);
}

// ---------------------------------------------------------------------------
// Host orchestration
// ---------------------------------------------------------------------------
extern "C" void kernel_launch(void* const* d_in, const int* in_sizes, int n_in,
                              void* d_out, int out_size, void* d_ws, size_t ws_size,
                              hipStream_t stream) {
  (void)in_sizes; (void)n_in; (void)out_size; (void)ws_size;
  constexpr size_t ROWS = 262144;   // B*T*N == B*N*T
  constexpr int    BN   = 4096;     // B*N
  constexpr int    BE   = 16384;    // B*E

  auto P = [&](int i) { return (const float*)d_in[i]; };
  const int* ei  = (const int*)d_in[4];
  const int* src = ei;
  const int* dst = ei + 2048;

  // param indices (jax pytree flatten: dicts sorted by key):
  // 5,6 edge_pool | 7..14 f2s{Wk,Wo,Wq,Wv,bk,bo,bq,bv} | 15,16 fast | 17,18 gate
  // 19,20 ln_fast | 21,22 ln_slow | 23,24 out | 25..32 s2f | 33,34 slow
  // 35..38 gat0{W,a_dst,a_src,bias} | 39..42 gat1 | 43,44 in_proj
  // 45..54 param_head | 55..66 pmu x3 | 67..74 scada x2 | 75..84 state_head

  float* w = (float*)d_ws;
  size_t off = 0;
  auto alloc = [&](size_t n) { float* p = w + off; off += n; return p; };
  float* hw    = alloc(ROWS * 256);
  float* aggr  = alloc(ROWS * 256);
  float* seq   = alloc(ROWS * 256);
  float* spmu  = alloc(ROWS * 256);
  float* sscd  = alloc(ROWS * 256);
  float* ybuf  = alloc(ROWS * 256);
  float* gates = alloc((size_t)BN * 1024);
  float* hst   = alloc((size_t)BN * 256);
  float* cst   = alloc((size_t)BN * 256);
  float* als   = alloc(ROWS * 4);
  float* ald   = alloc(ROWS * 4);
  float* mb    = alloc(ROWS * 4);
  float* sb    = alloc(ROWS * 4);
  float* hfast = alloc(ROWS * 128);
  float* hslow = alloc(ROWS * 128);
  float* qb    = alloc(ROWS * 128);
  float* kb    = alloc(ROWS * 128);
  float* vb    = alloc(ROWS * 128);
  float* ob    = alloc(ROWS * 128);
  float* afast = alloc(ROWS * 128);
  float* aslow = alloc(ROWS * 128);
  float* fpool = alloc((size_t)BN * 128);
  float* spool = alloc((size_t)BN * 128);
  float* gateb = alloc((size_t)BN * 128);
  float* fused = alloc((size_t)BN * 128);
  float* t128  = alloc((size_t)BE * 128);
  float* t64   = alloc((size_t)BE * 64);
  float* eh    = alloc((size_t)BE * 128);

  unsigned short* ub = (unsigned short*)(w + off);
  size_t uoff = 0;
  auto ualloc = [&](size_t n) { unsigned short* p = ub + uoff; uoff += n; return p; };
  unsigned short* h0bf    = ualloc(ROWS * 64);
  unsigned short* aggrbf  = ualloc(ROWS * 256);
  unsigned short* spmubf  = ualloc(ROWS * 256);
  unsigned short* sscdbf  = ualloc(ROWS * 256);
  unsigned short* hstbf   = ualloc((size_t)BN * 256);
  unsigned short* hfastbf = ualloc(ROWS * 128);
  unsigned short* hslowbf = ualloc(ROWS * 128);
  unsigned short* abf     = ualloc(ROWS * 128);
  unsigned short* catbf   = ualloc((size_t)BN * 256);
  unsigned short* cat2bf  = ualloc((size_t)BN * 256);
  unsigned short* fusedbf = ualloc((size_t)BN * 128);
  unsigned short* t128bf  = ualloc((size_t)BE * 128);
  unsigned short* t64bf   = ualloc((size_t)BE * 64);
  unsigned short* efbf    = ualloc((size_t)BE * 256);
  unsigned short* ehbf    = ualloc((size_t)BE * 128);

  auto wprep = [&](int pidx, int K, int N, int transpose) {
    unsigned short* dstw = ualloc((size_t)K * N);
    k_wT<<<(K * N + 255) / 256, 256, 0, stream>>>(P(pidx), dstw, K, N, transpose);
    return dstw;
  };
  unsigned short* wb_gat0 = wprep(35, 64, 256, 0);
  unsigned short* wb_gat1 = wprep(39, 256, 256, 0);
  unsigned short* wb_ih[5]; unsigned short* wb_hh[5];
  const int lstm_base[5] = {55, 59, 63, 67, 71};
  for (int l = 0; l < 5; l++) {
    wb_ih[l] = wprep(lstm_base[l],     256, 1024, 1);
    wb_hh[l] = wprep(lstm_base[l] + 1, 256, 1024, 1);
  }
  unsigned short* wb_slow = wprep(33, 256, 128, 1);
  unsigned short* wb_fast = wprep(15, 256, 128, 1);
  unsigned short* wb_f2s[4], *wb_s2f[4];       // Wk,Wo,Wq,Wv
  for (int j = 0; j < 4; j++) wb_f2s[j] = wprep(7 + j,  128, 128, 1);
  for (int j = 0; j < 4; j++) wb_s2f[j] = wprep(25 + j, 128, 128, 1);
  unsigned short* wb_gate = wprep(17, 256, 128, 1);
  unsigned short* wb_out  = wprep(23, 256, 128, 1);
  unsigned short* wb_ep   = wprep(5,  256, 128, 1);
  unsigned short* wb_sh0  = wprep(75, 128, 128, 1);
  unsigned short* wb_sh1  = wprep(79, 128, 64, 1);
  unsigned short* wb_sh2  = wprep(83, 64, 2, 1);
  unsigned short* wb_ph0  = wprep(45, 128, 128, 1);
  unsigned short* wb_ph1  = wprep(49, 128, 64, 1);
  unsigned short* wb_ph2  = wprep(53, 64, 2, 1);

  auto gemm = [&](const unsigned short* A, int lda, const unsigned short* Wb, int ldw,
                  float* C, int ldc, unsigned short* Cbf, const float* bias,
                  int M, int N, int K, int flags) {
    if ((M & 127) == 0 && M >= 16384) {
      dim3 g((N + 63) / 64, M / 128), b(128);
      k_gemm<2><<<g, b, 0, stream>>>(A, lda, Wb, ldw, C, ldc, Cbf, bias, M, N, K, flags);
    } else {
      dim3 g((N + 63) / 64, (M + 63) / 64), b(128);
      k_gemm<1><<<g, b, 0, stream>>>(A, lda, Wb, ldw, C, ldc, Cbf, bias, M, N, K, flags);
    }
  };
  auto fill = [&](float* p, size_t n, float v) {
    k_fill<<<(unsigned)((n + 255) / 256), 256, 0, stream>>>(p, n, v);
  };

  // ---- 1. input projection (K=4: scalar) ----
  k_input_proj<<<(unsigned)(ROWS * 64 / 256), 256, 0, stream>>>(
      P(0), P(1), P(2), P(3), P(43), P(44), h0bf);

  // ---- 2. GAT layers ----
  auto run_gat = [&](const unsigned short* inbf, int fin, const unsigned short* Wg,
                     const float* a_d, const float* a_s, const float* bias) {
    gemm(inbf, fin, Wg, 256, hw, 256, nullptr, nullptr, (int)ROWS, 256, fin, 0);
    k_gat_logits<<<4096, 256, 0, stream>>>(hw, a_s, a_d, als, ald);
    k_gat_maxsum<<<512, 256, 0, stream>>>(als, ald, src, dst, mb, sb);
    fill(aggr, ROWS * 256, 0.f);
    k_gat_aggregate<<<512 * 2048, 256, 0, stream>>>(hw, als, ald, mb, sb, src, dst, aggr);
    k_bias_elu<<<(unsigned)ROWS, 256, 0, stream>>>(aggr, bias, aggrbf);
  };
  run_gat(h0bf, 64, wb_gat0, P(36), P(37), P(38));
  run_gat(aggrbf, 256, wb_gat1, P(40), P(41), P(42));
  k_seq_tr<<<(unsigned)ROWS, 256, 0, stream>>>(aggr, seq);

  // ---- 3. residual LSTM streams ----
  (void)hipMemcpyAsync(spmu, seq, ROWS * 256 * sizeof(float), hipMemcpyDeviceToDevice, stream);
  (void)hipMemcpyAsync(sscd, seq, ROWS * 256 * sizeof(float), hipMemcpyDeviceToDevice, stream);
  k_cvt<<<(unsigned)(ROWS * 256 / 256), 256, 0, stream>>>(spmu, spmubf, ROWS * 256);
  k_cvt<<<(unsigned)(ROWS * 256 / 256), 256, 0, stream>>>(sscd, sscdbf, ROWS * 256);
  auto run_lstm = [&](float* S, unsigned short* Sbf, int l) {
    const float* bih = P(lstm_base[l] + 2), *bhh = P(lstm_base[l] + 3);
    fill(hst, (size_t)BN * 256, 0.f);
    fill(cst, (size_t)BN * 256, 0.f);
    k_fillu<<<(BN * 256 + 255) / 256, 256, 0, stream>>>(hstbf, (size_t)BN * 256);
    for (int t = 0; t < 64; t++) {
      gemm(Sbf + (size_t)t * 256, 64 * 256, wb_ih[l], 1024, gates, 1024, nullptr, nullptr,
           BN, 1024, 256, 0);
      gemm(hstbf, 256, wb_hh[l], 1024, gates, 1024, nullptr, nullptr, BN, 1024, 256, 1);
      k_lstm_point<<<BN, 256, 0, stream>>>(gates, bih, bhh, hst, hstbf, cst, ybuf, t);
    }
    k_add_sh<<<(unsigned)(ROWS * 256 / 256), 256, 0, stream>>>(S, ybuf, Sbf, ROWS * 256);
  };
  run_lstm(spmu, spmubf, 0); run_lstm(spmu, spmubf, 1); run_lstm(spmu, spmubf, 2);
  run_lstm(sscd, sscdbf, 3); run_lstm(sscd, sscdbf, 4);

  // ---- 4. fusion projections + cross attention ----
  gemm(sscdbf, 256, wb_slow, 128, hslow, 128, hslowbf, P(34), (int)ROWS, 128, 256, 0);
  gemm(spmubf, 256, wb_fast, 128, hfast, 128, hfastbf, P(16), (int)ROWS, 128, 256, 0);
  auto run_mha = [&](const unsigned short* qinbf, const unsigned short* kvinbf,
                     unsigned short** Wm, int mbase, const float* qres,
                     const float* lg, const float* lb, float* outb) {
    gemm(qinbf, 128, Wm[2], 128, qb, 128, nullptr, P(mbase + 6), (int)ROWS, 128, 128, 0);
    gemm(kvinbf, 128, Wm[0], 128, kb, 128, nullptr, P(mbase + 4), (int)ROWS, 128, 128, 0);
    gemm(kvinbf, 128, Wm[3], 128, vb, 128, nullptr, P(mbase + 7), (int)ROWS, 128, 128, 0);
    k_mha_attn<<<dim3(BN, 4), 64, 0, stream>>>(qb, kb, vb, abf);
    gemm(abf, 128, Wm[1], 128, ob, 128, nullptr, P(mbase + 5), (int)ROWS, 128, 128, 0);
    k_res_ln<<<(unsigned)ROWS, 128, 0, stream>>>(qres, ob, lg, lb, outb, nullptr, 128);
  };
  run_mha(hfastbf, hslowbf, wb_f2s, 7,  hfast, P(19), P(20), afast);
  run_mha(hslowbf, hfastbf, wb_s2f, 25, hslow, P(21), P(22), aslow);

  // ---- 5. pooling, gate, fused ----
  k_mean_pool<<<BN, 128, 0, stream>>>(afast, fpool);
  k_mean_pool<<<BN, 128, 0, stream>>>(aslow, spool);
  k_concat2<<<BN, 256, 0, stream>>>(fpool, spool, catbf);
  gemm(catbf, 256, wb_gate, 128, gateb, 128, nullptr, P(18), BN, 128, 256, 4 /*sigmoid*/);
  k_gated_cat<<<BN, 256, 0, stream>>>(fpool, spool, gateb, cat2bf);
  gemm(cat2bf, 256, wb_out, 128, fused, 128, fusedbf, P(24), BN, 128, 256, 0);

  // ---- 6. heads ----
  float* out = (float*)d_out;
  auto head_mlp = [&](const unsigned short* inbf, int rows,
                      const unsigned short* W0, const unsigned short* W1,
                      const unsigned short* W2, int pb, float* dst_out) {
    gemm(inbf, 128, W0, 128, t128, 128, nullptr, P(pb + 1), rows, 128, 128, 2 /*relu*/);
    k_res_ln<<<rows, 128, 0, stream>>>(t128, nullptr, P(pb + 2), P(pb + 3), t128, t128bf, 128);
    gemm(t128bf, 128, W1, 64, t64, 64, nullptr, P(pb + 5), rows, 64, 128, 2);
    k_res_ln<<<rows, 64, 0, stream>>>(t64, nullptr, P(pb + 6), P(pb + 7), t64, t64bf, 64);
    gemm(t64bf, 64, W2, 2, dst_out, 2, nullptr, P(pb + 9), rows, 2, 64, 0);
  };
  head_mlp(fusedbf, BN, wb_sh0, wb_sh1, wb_sh2, 75, out);          // state head

  k_ef_build<<<BE, 256, 0, stream>>>(fused, src, dst, efbf);
  gemm(efbf, 256, wb_ep, 128, eh, 128, ehbf, P(6), BE, 128, 256, 0);
  head_mlp(ehbf, BE, wb_ph0, wb_ph1, wb_ph2, 45, out + (size_t)BN * 2);  // param head
}
